// QuantizeEMA_53575422051165
// MI455X (gfx1250) — matrix operand order repaired and checked by reading, not compiled, not run on hardware
//
#include <hip/hip_runtime.h>
#include <hip/hip_bf16.h>

// ---------------------------------------------------------------------------
// VQ-VAE quantize + EMA update for MI455X (gfx1250, wave32, WMMA)
// inputs:(32,64,64,64) f32, embed:(64,512) f32, cluster_size:(512), embed_avg:(64,512)
// ---------------------------------------------------------------------------

typedef __attribute__((ext_vector_type(16))) __bf16 v16bf;
typedef __attribute__((ext_vector_type(8)))  __bf16 v8bf;
typedef __attribute__((ext_vector_type(8)))  float  v8f;

#define CDIM      64
#define HW        4096            // 64*64
#define N_TOT     131072          // 32*4096
#define NE        512
#define ROWS_PB   256             // rows (spatial positions) per block
#define XS_P      72              // LDS row pitch in bf16 elements
#define DECAYF    0.99f
#define EPSF      1e-5f

// bf16 round-to-nearest-even from f32
__device__ __forceinline__ unsigned short bf16_bits(float f) {
  unsigned int u = __float_as_uint(f);
  u += 0x7FFFu + ((u >> 16) & 1u);
  return (unsigned short)(u >> 16);
}

// A fragment (16-bit A 16x32 layout) from two aligned 8x-bf16 LDS chunks
__device__ __forceinline__ v16bf load_afrag(const unsigned short* rowp, int k1, int k2) {
  v8bf lo = *(const v8bf*)(rowp + k1);
  v8bf hi = *(const v8bf*)(rowp + k2);
  return __builtin_shufflevector(lo, hi, 0, 1, 2, 3, 4, 5, 6, 7,
                                 8, 9, 10, 11, 12, 13, 14, 15);
}

// ---------------------------------------------------------------------------
// Kernel 1: prep — swizzle embed into per-lane WMMA B fragments (bf16),
// compute enorm, zero counts+embed_sum scratch.
//  B fragment order used here (contiguous k, A matches): lanes 0-15 hold K=kk*32+0..15 (elem j -> K=j),
//  lanes 16-31 hold K=kk*32+16..31 ; N = lane & 15.
// ---------------------------------------------------------------------------
__global__ void vq_prep(const float* __restrict__ embed,
                        unsigned short* __restrict__ embedB,
                        float* __restrict__ enorm,
                        float* __restrict__ zbase /* counts ++ embed_sum */) {
  int gtid = blockIdx.x * 256 + threadIdx.x;           // 10 blocks * 256 = 2560
  if (gtid < 2048) {
    int t    = gtid >> 6;         // code tile 0..31
    int rem  = gtid & 63;
    int kk   = rem >> 5;          // K half 0..1
    int lane = rem & 31;
    int half = lane >> 4;
    int ncol = lane & 15;
    int code = t * 16 + ncol;
    int kstart = kk * 32 + half * 16;
    unsigned short* dst = embedB + (size_t)gtid * 16;
#pragma unroll
    for (int j = 0; j < 16; ++j)
      dst[j] = bf16_bits(embed[(size_t)(kstart + j) * NE + code]);
  } else if (gtid < 2560) {
    int code = gtid - 2048;
    float s = 0.0f;
#pragma unroll
    for (int k = 0; k < CDIM; ++k) {
      float e = embed[(size_t)k * NE + code];
      s += e * e;
    }
    enorm[code] = s;
  }
  for (int i = gtid; i < NE + CDIM * NE; i += 2560) zbase[i] = 0.0f;
}

// ---------------------------------------------------------------------------
// Kernel 2: main — distances via bf16 WMMA (2 row-tiles per wave, B reused),
// argmin, q/diff/ind outputs, EMA scatter.
// grid = 512 blocks, 256 threads (8 waves); block: 256 rows x all 512 codes.
// ---------------------------------------------------------------------------
__global__ void __launch_bounds__(256)
vq_main(const float* __restrict__ inputs,
        const float* __restrict__ embed,
        const unsigned short* __restrict__ embedB,
        const float* __restrict__ enorm,
        float* __restrict__ q_out,
        float* __restrict__ diff_out,
        float* __restrict__ ind_out,
        float* __restrict__ counts,
        float* __restrict__ embed_sum) {
  __shared__ unsigned short xsb[ROWS_PB * XS_P];   // x tile as bf16 bits (36 KB)
  __shared__ float en_s[NE];
  __shared__ int   inds_s[ROWS_PB];

  const int tid = threadIdx.x;
  const int n0  = blockIdx.x << 8;                 // first row of this block
  const int b   = n0 >> 12;                        // image index (4096 rows / image)
  const int hw0 = n0 & 4095;
  const size_t obase = (size_t)b * (CDIM * HW) + hw0;

  // ---- stage x tile (256 rows x 64 ch) as bf16: coalesced loads per channel ----
#pragma unroll
  for (int i = 0; i < 64; ++i) {
    int lin = tid + i * 256;
    int c = lin >> 8;
    int r = lin & 255;
    xsb[r * XS_P + c] = bf16_bits(inputs[obase + (size_t)c * HW + r]);
  }
  en_s[tid]       = enorm[tid];
  en_s[tid + 256] = enorm[tid + 256];
  __syncthreads();

  // ---- per-wave WMMA: 32 rows (two 16-row tiles) x 512 codes ----
  const int wave = tid >> 5;
  const int lane = tid & 31;
  const int half = lane >> 4;
  const int mrow = lane & 15;
  const int wr   = wave * 32;                      // wave's first row in tile

  // A fragments: lanes0-15 K 0..15 ; lanes16-31 K 16..31
  const unsigned short* rowa = &xsb[(wr + mrow) * XS_P];
  const unsigned short* rowb = &xsb[(wr + 16 + mrow) * XS_P];
  const int k1 = half * 16;
  v16bf Aa0 = load_afrag(rowa, k1,      8 + k1);
  v16bf Aa1 = load_afrag(rowa, 32 + k1, 40 + k1);
  v16bf Ab0 = load_afrag(rowb, k1,      8 + k1);
  v16bf Ab1 = load_afrag(rowb, 32 + k1, 40 + k1);

  float bestv[16];
  int   besti[16];
#pragma unroll
  for (int g = 0; g < 16; ++g) { bestv[g] = 3.4e38f; besti[g] = 0; }

  const v16bf* Bp = (const v16bf*)embedB;
  for (int t = 0; t < 32; ++t) {
    v16bf B0 = Bp[(t * 2 + 0) * 32 + lane];
    v16bf B1 = Bp[(t * 2 + 1) * 32 + lane];
    int   code = t * 16 + mrow;                    // this lane's N column
    float en = en_s[code];

    // argmin is invariant to the per-row ||x||^2 constant: d' = ||e||^2 - 2 x.e
    v8f acca = {0.f, 0.f, 0.f, 0.f, 0.f, 0.f, 0.f, 0.f};
    acca = __builtin_amdgcn_wmma_f32_16x16x32_bf16(false, Aa0, false, B0,
                                                   (short)0, acca, false, false);
    acca = __builtin_amdgcn_wmma_f32_16x16x32_bf16(false, Aa1, false, B1,
                                                   (short)0, acca, false, false);
#pragma unroll
    for (int g = 0; g < 8; ++g) {                  // D: VGPR g -> M = half*8+g
      float d = __builtin_fmaf(-2.0f, acca[g], en);
      if (d < bestv[g]) { bestv[g] = d; besti[g] = code; } // strict <: first-min
    }

    v8f accb = {0.f, 0.f, 0.f, 0.f, 0.f, 0.f, 0.f, 0.f};
    accb = __builtin_amdgcn_wmma_f32_16x16x32_bf16(false, Ab0, false, B0,
                                                   (short)0, accb, false, false);
    accb = __builtin_amdgcn_wmma_f32_16x16x32_bf16(false, Ab1, false, B1,
                                                   (short)0, accb, false, false);
#pragma unroll
    for (int g = 0; g < 8; ++g) {
      float d = __builtin_fmaf(-2.0f, accb[g], en);
      if (d < bestv[8 + g]) { bestv[8 + g] = d; besti[8 + g] = code; }
    }
  }

  // cross-lane argmin within each 16-lane half (ties -> lowest index, like jnp.argmin)
#pragma unroll
  for (int q = 0; q < 16; ++q) {
    float v = bestv[q];
    int   idx = besti[q];
#pragma unroll
    for (int off = 8; off >= 1; off >>= 1) {
      float ov = __shfl_xor(v, off, 32);
      int   oi = __shfl_xor(idx, off, 32);
      if (ov < v || (ov == v && oi < idx)) { v = ov; idx = oi; }
    }
    if (mrow == 0) inds_s[wr + (q >> 3) * 16 + half * 8 + (q & 7)] = idx;
  }
  __syncthreads();

  // ---- outputs: q (=quantized = straight-through), diff, ind + EMA scatter ----
#pragma unroll 4
  for (int i = 0; i < 64; ++i) {
    int lin = tid + i * 256;
    int c = lin >> 8;
    int r = lin & 255;
    int ind = inds_s[r];
    size_t o = obase + (size_t)c * HW + r;          // coalesced
    float q = embed[(size_t)c * NE + ind];          // L2-hot (128 KB)
    float x = inputs[o];                            // re-read f32 (L2-hot)
    q_out[o]    = q;
    diff_out[o] = q - x;
    atomicAdd(&embed_sum[(size_t)c * NE + ind], x);
  }
  ind_out[n0 + tid] = (float)inds_s[tid];
  atomicAdd(&counts[inds_s[tid]], 1.0f);
}

// ---------------------------------------------------------------------------
// Kernel 3: finalize EMA — new_cluster_size, new_embed_avg, new_embed
// ---------------------------------------------------------------------------
__global__ void vq_finalize(const float* __restrict__ cluster_size,
                            const float* __restrict__ embed_avg,
                            const float* __restrict__ counts,
                            const float* __restrict__ embed_sum,
                            float* __restrict__ ne_out,
                            float* __restrict__ ncs_out,
                            float* __restrict__ nea_out) {
  __shared__ float red[NE];
  int j = threadIdx.x;                              // 512 threads
  float ncs = cluster_size[j] * DECAYF + counts[j] * (1.0f - DECAYF);
  ncs_out[j] = ncs;
  red[j] = ncs;
  __syncthreads();
  for (int s = 256; s > 0; s >>= 1) {
    if (j < s) red[j] += red[j + s];
    __syncthreads();
  }
  float n  = red[0];
  float cs = (ncs + EPSF) / (n + (float)NE * EPSF) * n;
#pragma unroll 4
  for (int c = 0; c < CDIM; ++c) {
    float nea = embed_avg[(size_t)c * NE + j] * DECAYF +
                embed_sum[(size_t)c * NE + j] * (1.0f - DECAYF);
    nea_out[(size_t)c * NE + j] = nea;
    ne_out[(size_t)c * NE + j]  = nea / cs;
  }
}

// ---------------------------------------------------------------------------
extern "C" void kernel_launch(void* const* d_in, const int* in_sizes, int n_in,
                              void* d_out, int out_size, void* d_ws, size_t ws_size,
                              hipStream_t stream) {
  (void)in_sizes; (void)n_in; (void)out_size; (void)ws_size;

  const float* inputs       = (const float*)d_in[0];   // (32,64,64,64)
  const float* embed        = (const float*)d_in[1];   // (64,512)
  const float* cluster_size = (const float*)d_in[2];   // (512)
  const float* embed_avg    = (const float*)d_in[3];   // (64,512)

  float* out      = (float*)d_out;
  float* q_out    = out;                               //  8388608
  float* diff_out = out + 8388608;                     //  8388608
  float* ind_out  = out + 16777216;                    //   131072
  float* ne_out   = out + 16908288;                    //    32768
  float* ncs_out  = out + 16941056;                    //      512
  float* nea_out  = out + 16941568;                    //    32768

  // workspace layout
  unsigned short* embedB = (unsigned short*)d_ws;                       // 65536 B
  float* enorm     = (float*)((char*)d_ws + 65536);                     //  2048 B
  float* counts    = (float*)((char*)d_ws + 65536 + 2048);              //  2048 B
  float* embed_sum = (float*)((char*)d_ws + 65536 + 4096);              // 131072 B

  vq_prep<<<10, 256, 0, stream>>>(embed, embedB, enorm, counts /* zbase */);
  vq_main<<<N_TOT / ROWS_PB, 256, 0, stream>>>(inputs, embed, embedB, enorm,
                                               q_out, diff_out, ind_out,
                                               counts, embed_sum);
  vq_finalize<<<1, NE, 0, stream>>>(cluster_size, embed_avg, counts, embed_sum,
                                    ne_out, ncs_out, nea_out);
}
